// TNetEdge_41549513621913
// MI455X (gfx1250) — matrix operand order repaired and checked by reading, not compiled, not run on hardware
//
#include <hip/hip_runtime.h>
#include <hip/hip_bf16.h>
#include <float.h>

#define B_   16
#define N_   2048
#define K_   20
#define EPS_ 1e-3f

typedef __attribute__((ext_vector_type(16))) _Float16 v16h;
typedef __attribute__((ext_vector_type(8)))  float    v8f;

// ---------------------------------------------------------------------------
// prep: pad C-dim features to 64, cast to f16, compute squared norms (f32)
// ---------------------------------------------------------------------------
template <int CIN>
__global__ void prep_kernel(const float* __restrict__ X,
                            _Float16* __restrict__ xh,
                            float* __restrict__ sqn) {
    int p = blockIdx.x * blockDim.x + threadIdx.x;
    if (p >= B_ * N_) return;
    const float* xr = X + (size_t)p * CIN;
    _Float16* hr = xh + (size_t)p * 64;
    float s = 0.f;
#pragma unroll
    for (int c = 0; c < 64; ++c) {
        float v = (c < CIN) ? xr[c] : 0.f;
        hr[c] = (_Float16)v;
        s += v * v;
    }
    sqn[p] = s;
}

// ---------------------------------------------------------------------------
// knn: one wave per 16-row block. Gram matrix tiles via v_wmma_f32_16x16x32_f16,
// fused register-resident top-20 (smallest distance) per row.
// ---------------------------------------------------------------------------
__global__ void __launch_bounds__(32)
knn_kernel(const _Float16* __restrict__ xh,
           const float* __restrict__ sqn,
           int* __restrict__ nn) {
    __shared__ float tile[16][16];
    const int lane  = threadIdx.x;
    const int b     = blockIdx.y;
    const int rbase = blockIdx.x * 16;
    const int hi    = lane >> 4;   // half-wave select
    const int lo    = lane & 15;

    // A fragment: 16 rows (this block) x K=64, per ISA 16-bit A 16x32 layout:
    // VGPR v: lanes0-15 K = (v&4)*4 + 2*(v&3); lanes16-31 add +8.
    const _Float16* xrow = xh + ((size_t)b * N_ + rbase + lo) * 64;
    v16h a0, a1;
#pragma unroll
    for (int v = 0; v < 8; ++v) {
        int k0 = ((v & 4) << 2) + (hi << 3) + ((v & 3) << 1);
        a0[2 * v]     = xrow[k0];
        a0[2 * v + 1] = xrow[k0 + 1];
        a1[2 * v]     = xrow[32 + k0];
        a1[2 * v + 1] = xrow[32 + k0 + 1];
    }
    float sqr[8];
#pragma unroll
    for (int v = 0; v < 8; ++v)
        sqr[v] = sqn[b * N_ + rbase + v + hi * 8];

    float bd[K_]; int bi[K_];
#pragma unroll
    for (int j = 0; j < K_; ++j) { bd[j] = FLT_MAX; bi[j] = 0; }

    for (int ct = 0; ct < N_ / 16; ++ct) {
        const int col = ct * 16 + lo;
        // B fragment: 32x16, same K order as A: VGPR v: lanes0-15 K = (v&4)*4 + 2*(v&3); lanes16-31 add +8.
        const _Float16* xc = xh + ((size_t)b * N_ + col) * 64;
        v16h bf0, bf1;
#pragma unroll
        for (int v = 0; v < 8; ++v) {
            int kb = ((v & 4) << 2) + (hi << 3) + ((v & 3) << 1);
            bf0[2 * v]     = xc[kb];
            bf0[2 * v + 1] = xc[kb + 1];
            bf1[2 * v]     = xc[32 + kb];
            bf1[2 * v + 1] = xc[32 + kb + 1];
        }
        v8f acc = {0.f, 0.f, 0.f, 0.f, 0.f, 0.f, 0.f, 0.f};
        acc = __builtin_amdgcn_wmma_f32_16x16x32_f16(false, a0, false, bf0,
                                                     (short)0, acc, false, false);
        acc = __builtin_amdgcn_wmma_f32_16x16x32_f16(false, a1, false, bf1,
                                                     (short)0, acc, false, false);
        float sqc = sqn[b * N_ + col];
#pragma unroll
        for (int v = 0; v < 8; ++v)
            tile[v + hi * 8][lo] = sqr[v] - 2.f * acc[v] + sqc;
        __syncthreads();
        if (lane < 16) {
#pragma unroll
            for (int c = 0; c < 16; ++c) {
                float dcand = tile[lane][c];
                int idx = ct * 16 + c;
                if (dcand < bd[K_ - 1]) {
                    bd[K_ - 1] = dcand; bi[K_ - 1] = idx;
#pragma unroll
                    for (int j = K_ - 1; j > 0; --j) {
                        if (bd[j] < bd[j - 1]) {
                            float td = bd[j]; bd[j] = bd[j - 1]; bd[j - 1] = td;
                            int   ti = bi[j]; bi[j] = bi[j - 1]; bi[j - 1] = ti;
                        }
                    }
                }
            }
        }
        __syncthreads();
    }
    if (lane < 16) {
        int* row = nn + ((size_t)b * N_ + rbase + lane) * K_;
#pragma unroll
        for (int j = 0; j < K_; ++j) row[j] = bi[j];
    }
}

// ---------------------------------------------------------------------------
// edgeconv: wave per point, 2 output features per lane, W staged in LDS,
// LN via shuffle reduction, relu, max over k neighbors.
// ---------------------------------------------------------------------------
template <int CIN>
__global__ void __launch_bounds__(256)
edgeconv_kernel(const float* __restrict__ X, const int* __restrict__ nn,
                const float* __restrict__ W, const float* __restrict__ g,
                const float* __restrict__ bt, float* __restrict__ Y) {
    __shared__ float Ws[2 * CIN * 64];
    __shared__ float gs[64], bs[64];
    const int tid = threadIdx.x;
    for (int i = tid; i < 2 * CIN * 64; i += 256) Ws[i] = W[i];
    if (tid < 64) { gs[tid] = g[tid]; bs[tid] = bt[tid]; }
    __syncthreads();

    const int wave = tid >> 5;
    const int lane = tid & 31;
    const int p = blockIdx.x * 8 + wave;      // global point id
    const int b = p / N_;
    const float* xn = X + (size_t)p * CIN;

    // common term: x_n * W_top
    float t0 = 0.f, t1 = 0.f;
#pragma unroll 4
    for (int c = 0; c < CIN; ++c) {
        float xc = xn[c];
        t0 += xc * Ws[c * 64 + lane];
        t1 += xc * Ws[c * 64 + lane + 32];
    }
    float m0 = 0.f, m1 = 0.f;   // relu output >= 0, so 0 is a valid identity for max
    const int* nrow = nn + (size_t)p * K_;
    for (int j = 0; j < K_; ++j) {
        const float* xm = X + ((size_t)b * N_ + nrow[j]) * CIN;
        float h0 = t0, h1 = t1;
#pragma unroll 4
        for (int c = 0; c < CIN; ++c) {
            float dc = xm[c] - xn[c];
            h0 += dc * Ws[(CIN + c) * 64 + lane];
            h1 += dc * Ws[(CIN + c) * 64 + lane + 32];
        }
        float s = h0 + h1;
#pragma unroll
        for (int o = 16; o > 0; o >>= 1) s += __shfl_xor(s, o, 32);
        float mean = s * (1.f / 64.f);
        float d0 = h0 - mean, d1 = h1 - mean;
        float s2 = d0 * d0 + d1 * d1;
#pragma unroll
        for (int o = 16; o > 0; o >>= 1) s2 += __shfl_xor(s2, o, 32);
        float r = rsqrtf(s2 * (1.f / 64.f) + EPS_);
        float v0 = fmaxf(d0 * r * gs[lane] + bs[lane], 0.f);
        float v1 = fmaxf(d1 * r * gs[lane + 32] + bs[lane + 32], 0.f);
        m0 = fmaxf(m0, v0);
        m1 = fmaxf(m1, v1);
    }
    float* yr = Y + (size_t)p * 64;
    yr[lane] = m0;
    yr[lane + 32] = m1;
}

// ---------------------------------------------------------------------------
// zero-fill helper (pool buffer must start at 0 since ws is poisoned)
// ---------------------------------------------------------------------------
__global__ void zero_kernel(float* __restrict__ p, int n) {
    int i = blockIdx.x * blockDim.x + threadIdx.x;
    if (i < n) p[i] = 0.f;
}

// ---------------------------------------------------------------------------
// convpool: 64->512 conv + LN + relu, then global max over points via
// int-bitcast atomicMax (valid: values are relu'd >= 0, pool init 0).
// ---------------------------------------------------------------------------
__global__ void __launch_bounds__(256)
convpool_kernel(const float* __restrict__ X, const float* __restrict__ W,
                const float* __restrict__ g, const float* __restrict__ bt,
                float* __restrict__ pool) {
    const int tid = threadIdx.x;
    const int wave = tid >> 5, lane = tid & 31;
    const int p = blockIdx.x * 8 + wave;
    const int b = p / N_;
    const float* xr = X + (size_t)p * 64;
    float acc[16];
#pragma unroll
    for (int t = 0; t < 16; ++t) acc[t] = 0.f;
    for (int c = 0; c < 64; ++c) {
        float xc = xr[c];
        const float* wr = W + (size_t)c * 512;
#pragma unroll
        for (int t = 0; t < 16; ++t)
            acc[t] += xc * wr[lane + 32 * t];
    }
    float s = 0.f;
#pragma unroll
    for (int t = 0; t < 16; ++t) s += acc[t];
#pragma unroll
    for (int o = 16; o > 0; o >>= 1) s += __shfl_xor(s, o, 32);
    float mean = s * (1.f / 512.f);
    float s2 = 0.f;
#pragma unroll
    for (int t = 0; t < 16; ++t) { float d = acc[t] - mean; s2 += d * d; }
#pragma unroll
    for (int o = 16; o > 0; o >>= 1) s2 += __shfl_xor(s2, o, 32);
    float r = rsqrtf(s2 * (1.f / 512.f) + EPS_);
    int* ip = (int*)pool + (size_t)b * 512;
#pragma unroll
    for (int t = 0; t < 16; ++t) {
        int f = lane + 32 * t;
        float v = fmaxf((acc[t] - mean) * r * g[f] + bt[f], 0.f);
        atomicMax(ip + f, __float_as_int(v));
    }
}

// ---------------------------------------------------------------------------
// head: per-batch dense 512->256->128->(3x3) with LN+relu, bias = identity.
// ---------------------------------------------------------------------------
__global__ void __launch_bounds__(256)
head_kernel(const float* __restrict__ pool,
            const float* __restrict__ Wd0, const float* __restrict__ gd0, const float* __restrict__ bd0,
            const float* __restrict__ Wd1, const float* __restrict__ gd1, const float* __restrict__ bd1,
            const float* __restrict__ Wt,  const float* __restrict__ btb,
            float* __restrict__ trans) {
    __shared__ float xp[512];
    __shared__ float h0[256];
    __shared__ float h1[128];
    __shared__ float red[256];
    __shared__ float stats[2];
    const int t = threadIdx.x;
    const int b = blockIdx.x;
    xp[t]       = pool[b * 512 + t];
    xp[t + 256] = pool[b * 512 + t + 256];
    __syncthreads();

    // dense 512 -> 256
    float a = 0.f;
    for (int c = 0; c < 512; ++c) a += xp[c] * Wd0[c * 256 + t];
    red[t] = a; __syncthreads();
    for (int o = 128; o > 0; o >>= 1) { if (t < o) red[t] += red[t + o]; __syncthreads(); }
    if (t == 0) stats[0] = red[0] * (1.f / 256.f);
    __syncthreads();
    float d = a - stats[0];
    red[t] = d * d; __syncthreads();
    for (int o = 128; o > 0; o >>= 1) { if (t < o) red[t] += red[t + o]; __syncthreads(); }
    if (t == 0) stats[1] = rsqrtf(red[0] * (1.f / 256.f) + EPS_);
    __syncthreads();
    h0[t] = fmaxf(d * stats[1] * gd0[t] + bd0[t], 0.f);
    __syncthreads();

    // dense 256 -> 128
    float a1 = 0.f;
    if (t < 128)
        for (int c = 0; c < 256; ++c) a1 += h0[c] * Wd1[c * 128 + t];
    red[t] = (t < 128) ? a1 : 0.f; __syncthreads();
    for (int o = 128; o > 0; o >>= 1) { if (t < o) red[t] += red[t + o]; __syncthreads(); }
    if (t == 0) stats[0] = red[0] * (1.f / 128.f);
    __syncthreads();
    float d1 = a1 - stats[0];
    red[t] = (t < 128) ? d1 * d1 : 0.f; __syncthreads();
    for (int o = 128; o > 0; o >>= 1) { if (t < o) red[t] += red[t + o]; __syncthreads(); }
    if (t == 0) stats[1] = rsqrtf(red[0] * (1.f / 128.f) + EPS_);
    __syncthreads();
    if (t < 128) h1[t] = fmaxf(d1 * stats[1] * gd1[t] + bd1[t], 0.f);
    __syncthreads();

    // transform head 128 -> 9, bias is flattened identity
    if (t < 9) {
        float a2 = btb[t];
        for (int c = 0; c < 128; ++c) a2 += h1[c] * Wt[c * 9 + t];
        trans[b * 9 + t] = a2;
    }
}

// ---------------------------------------------------------------------------
// apply: out[b,n,:] = points[b,n,:] @ T[b]
// ---------------------------------------------------------------------------
__global__ void apply_kernel(const float* __restrict__ pts,
                             const float* __restrict__ trans,
                             float* __restrict__ out) {
    int p = blockIdx.x * blockDim.x + threadIdx.x;
    if (p >= B_ * N_) return;
    int b = p / N_;
    const float* T = trans + b * 9;
    float x = pts[p * 3], y = pts[p * 3 + 1], z = pts[p * 3 + 2];
    out[p * 3 + 0] = x * T[0] + y * T[3] + z * T[6];
    out[p * 3 + 1] = x * T[1] + y * T[4] + z * T[7];
    out[p * 3 + 2] = x * T[2] + y * T[5] + z * T[8];
}

// ---------------------------------------------------------------------------
extern "C" void kernel_launch(void* const* d_in, const int* in_sizes, int n_in,
                              void* d_out, int out_size, void* d_ws, size_t ws_size,
                              hipStream_t stream) {
    (void)in_sizes; (void)n_in; (void)out_size; (void)ws_size;
    const float* points = (const float*)d_in[0];
    // d_in[1] = k (constant 20, hardcoded)
    const float* W_e0 = (const float*)d_in[2];
    const float* g_e0 = (const float*)d_in[3];
    const float* b_e0 = (const float*)d_in[4];
    const float* W_e1 = (const float*)d_in[5];
    const float* g_e1 = (const float*)d_in[6];
    const float* b_e1 = (const float*)d_in[7];
    const float* W_e2 = (const float*)d_in[8];
    const float* g_e2 = (const float*)d_in[9];
    const float* b_e2 = (const float*)d_in[10];
    const float* W_c0 = (const float*)d_in[11];
    const float* g_c0 = (const float*)d_in[12];
    const float* b_c0 = (const float*)d_in[13];
    const float* W_d0 = (const float*)d_in[14];
    const float* g_d0 = (const float*)d_in[15];
    const float* b_d0 = (const float*)d_in[16];
    const float* W_d1 = (const float*)d_in[17];
    const float* g_d1 = (const float*)d_in[18];
    const float* b_d1 = (const float*)d_in[19];
    const float* W_t  = (const float*)d_in[20];
    const float* b_t  = (const float*)d_in[21];

    char* ws = (char*)d_ws;
    _Float16* xh  = (_Float16*)(ws);                     // 4,194,304 B
    float*    sqn = (float*)(ws + 4194304);              //   131,072 B
    int*      nn  = (int*)(ws + 4325376);                // 2,621,440 B
    float*    fa  = (float*)(ws + 6946816);              // 8,388,608 B
    float*    fb  = (float*)(ws + 15335424);             // 8,388,608 B
    float*    pool= (float*)(ws + 23724032);             //    32,768 B
    float*    tr  = (float*)(ws + 23756800);             //       576 B

    const dim3 knn_grid(N_ / 16, B_);

    // ---- EdgeConv 1 (input D=3) ----
    prep_kernel<3><<<128, 256, 0, stream>>>(points, xh, sqn);
    knn_kernel<<<knn_grid, 32, 0, stream>>>(xh, sqn, nn);
    edgeconv_kernel<3><<<B_ * N_ / 8, 256, 0, stream>>>(points, nn, W_e0, g_e0, b_e0, fa);

    // ---- EdgeConv 2 (64-d features) ----
    prep_kernel<64><<<128, 256, 0, stream>>>(fa, xh, sqn);
    knn_kernel<<<knn_grid, 32, 0, stream>>>(xh, sqn, nn);
    edgeconv_kernel<64><<<B_ * N_ / 8, 256, 0, stream>>>(fa, nn, W_e1, g_e1, b_e1, fb);

    // ---- EdgeConv 3 ----
    prep_kernel<64><<<128, 256, 0, stream>>>(fb, xh, sqn);
    knn_kernel<<<knn_grid, 32, 0, stream>>>(xh, sqn, nn);
    edgeconv_kernel<64><<<B_ * N_ / 8, 256, 0, stream>>>(fb, nn, W_e2, g_e2, b_e2, fa);

    // ---- conv 64->512 + LN + relu + global max pool ----
    zero_kernel<<<(B_ * 512 + 255) / 256, 256, 0, stream>>>(pool, B_ * 512);
    convpool_kernel<<<B_ * N_ / 8, 256, 0, stream>>>(fa, W_c0, g_c0, b_c0, pool);

    // ---- dense head -> 3x3 transforms ----
    head_kernel<<<B_, 256, 0, stream>>>(pool, W_d0, g_d0, b_d0,
                                        W_d1, g_d1, b_d1, W_t, b_t, tr);

    // ---- apply transform ----
    apply_kernel<<<128, 256, 0, stream>>>(points, tr, (float*)d_out);
}
